// GraphLayoutExtractor_51823075393752
// MI455X (gfx1250) — matrix operand order repaired and checked by reading, not compiled, not run on hardware
//
#include <hip/hip_runtime.h>

#define NODES   1024
#define EDGES   8192
#define NGRAPHS 8

typedef __attribute__((ext_vector_type(16))) __bf16 v16bf;
typedef __attribute__((ext_vector_type(8)))  float  v8f;

__device__ __forceinline__ __bf16 tobf(float f) { return (__bf16)f; }

// ---------------------------------------------------------------------------
// utility / elementwise kernels
// ---------------------------------------------------------------------------
__global__ void zero_f32(float* __restrict__ p, long n) {
    long i = (long)blockIdx.x * blockDim.x + threadIdx.x;
    if (i < n) p[i] = 0.f;
}

__global__ void count_deg(const int* __restrict__ dst, float* __restrict__ deg) {
    int e = blockIdx.x * blockDim.x + threadIdx.x;
    if (e < EDGES) atomicAdd(&deg[dst[e]], 1.f);
}

__global__ void gather_emb(const int* __restrict__ x, const float* __restrict__ emb,
                           float* __restrict__ out) {
    int i = blockIdx.x * blockDim.x + threadIdx.x;
    if (i < NODES * 64) {
        int n = i >> 6, c = i & 63;
        out[i] = emb[x[n] * 64 + c];
    }
}

// EF[e][k] (lda = Kp, zero padded): k<Cin: X[src][k]; k<Cin+3: pos[src]-pos[dst]; else 0
__global__ void build_ef(const float* __restrict__ X, int Cin, int Kp,
                         const float* __restrict__ pos,
                         const int* __restrict__ src, const int* __restrict__ dst,
                         float* __restrict__ EF) {
    long t = (long)blockIdx.x * blockDim.x + threadIdx.x;
    if (t >= (long)EDGES * Kp) return;
    int e = (int)(t / Kp), k = (int)(t % Kp);
    float v = 0.f;
    if (k < Cin) v = X[(long)src[e] * Cin + k];
    else if (k < Cin + 3) { int d = k - Cin; v = pos[src[e] * 3 + d] - pos[dst[e] * 3 + d]; }
    EF[t] = v;
}

__global__ void div_rows(float* __restrict__ m, const float* __restrict__ deg,
                         long total, int C) {
    long i = (long)blockIdx.x * blockDim.x + threadIdx.x;
    if (i < total) m[i] /= fmaxf(deg[(int)(i / C)], 1.f);
}

__global__ void edge_scatter_vec(const int* __restrict__ src, const int* __restrict__ dst,
                                 const float* __restrict__ X, float* __restrict__ acc, int D) {
    long t = (long)blockIdx.x * blockDim.x + threadIdx.x;
    if (t >= (long)EDGES * D) return;
    int e = (int)(t / D), i = (int)(t % D);
    atomicAdd(&acc[(long)dst[e] * D + i], X[(long)src[e] * D + i]);
}

// ---------------------------------------------------------------------------
// weight packing into WMMA B-fragment order (bf16, zero padded to ktiles*32)
// layout: [(ntile*ktiles + kt)*512 + lane*16 + e]
// ---------------------------------------------------------------------------
__global__ void pack_w_gemm(const float* __restrict__ W, int K, int N, int ktiles,
                            __bf16* __restrict__ out) {
    long t = (long)blockIdx.x * blockDim.x + threadIdx.x;
    long total = (long)(N / 16) * ktiles * 512;
    if (t >= total) return;
    int e = (int)(t & 15);
    int lane = (int)((t >> 4) & 31);
    long f = t >> 9;
    int kt = (int)(f % ktiles), nt = (int)(f / ktiles);
    int l15 = lane & 15, half = lane >> 4;
    int kb = kt * 32 + half * 8 + (e & 7) + (e >> 3) * 16;
    int col = nt * 16 + l15;
    out[t] = (kb < K) ? (__bf16)W[(long)kb * N + col] : (__bf16)0.f;
}

// conv weights (OIHW) with implicit-GEMM K flattened as k = tap*Cin + c
__global__ void pack_w_conv(const float* __restrict__ Wt, int Cin, int cinShift,
                            int KH, int KW, int Cout, int ktiles,
                            __bf16* __restrict__ out) {
    long t = (long)blockIdx.x * blockDim.x + threadIdx.x;
    long total = (long)(Cout / 16) * ktiles * 512;
    if (t >= total) return;
    int e = (int)(t & 15);
    int lane = (int)((t >> 4) & 31);
    long f = t >> 9;
    int kt = (int)(f % ktiles), nt = (int)(f / ktiles);
    int l15 = lane & 15, half = lane >> 4;
    int kb = kt * 32 + half * 8 + (e & 7) + (e >> 3) * 16;
    int tap = kb >> cinShift, c = kb & (Cin - 1);
    int taps = KH * KW;
    int col = nt * 16 + l15;
    float v = 0.f;
    if (tap < taps)
        v = Wt[(((long)col * Cin + c) * KH + tap / KW) * KW + (tap % KW)];
    out[t] = (__bf16)v;
}

// ---------------------------------------------------------------------------
// batch norm
// ---------------------------------------------------------------------------
__global__ void bn1d_stats(const float* __restrict__ X, int C,
                           float* __restrict__ mu, float* __restrict__ var) {
    __shared__ float s1[256], s2[256];
    int c = blockIdx.x, tid = threadIdx.x;
    float a = 0.f, b = 0.f;
    for (int i = tid; i < NODES; i += 256) {
        float v = X[(long)i * C + c];
        a += v; b += v * v;
    }
    s1[tid] = a; s2[tid] = b; __syncthreads();
    for (int s = 128; s > 0; s >>= 1) {
        if (tid < s) { s1[tid] += s1[tid + s]; s2[tid] += s2[tid + s]; }
        __syncthreads();
    }
    if (tid == 0) {
        float m = s1[0] / (float)NODES;
        mu[c] = m;
        var[c] = s2[0] / (float)NODES - m * m;
    }
}

__global__ void bn1d_apply(float* __restrict__ X, int C,
                           const float* __restrict__ mu, const float* __restrict__ var,
                           const float* __restrict__ g, const float* __restrict__ b) {
    int i = blockIdx.x * blockDim.x + threadIdx.x;
    if (i >= NODES * C) return;
    int c = i % C;
    float v = (X[i] - mu[c]) * rsqrtf(var[c] + 1e-5f) * g[c] + b[c];
    X[i] = v > 0.f ? v : 0.f;
}

// stats per channel; nhwc selects layout
__global__ void bn2d_stats(const float* __restrict__ X, int C, int S, int nhwc,
                           float* __restrict__ mu, float* __restrict__ var) {
    __shared__ float s1[256], s2[256];
    int c = blockIdx.x, tid = threadIdx.x;
    float a = 0.f, b = 0.f;
    long cnt = (long)NODES * S;
    for (long i = tid; i < cnt; i += 256) {
        long n = i / S, s = i % S;
        float v = nhwc ? X[(n * S + s) * C + c] : X[(n * C + c) * (long)S + s];
        a += v; b += v * v;
    }
    s1[tid] = a; s2[tid] = b; __syncthreads();
    for (int s = 128; s > 0; s >>= 1) {
        if (tid < s) { s1[tid] += s1[tid + s]; s2[tid] += s2[tid + s]; }
        __syncthreads();
    }
    if (tid == 0) {
        float m = s1[0] / (float)cnt;
        mu[c] = m;
        var[c] = s2[0] / (float)cnt - m * m;
    }
}

__global__ void make_scale_shift(const float* __restrict__ mu, const float* __restrict__ var,
                                 const float* __restrict__ g, const float* __restrict__ b,
                                 float* __restrict__ scale, float* __restrict__ shift, int C) {
    int c = blockIdx.x * blockDim.x + threadIdx.x;
    if (c < C) {
        float s = g[c] * rsqrtf(var[c] + 1e-5f);
        scale[c] = s;
        shift[c] = b[c] - mu[c] * s;
    }
}

// hh[n][y][x][c] (NHWC) = relu(bn(X_nchw[n][c][y/2][x/2]))
__global__ void up2_bn_relu(const float* __restrict__ X, float* __restrict__ Y,
                            int C, int SH, int SW,
                            const float* __restrict__ scale, const float* __restrict__ shift) {
    int OH = 2 * SH, OW = 2 * SW;
    long total = (long)NODES * OH * OW * C;
    long i = (long)blockIdx.x * blockDim.x + threadIdx.x;
    if (i >= total) return;
    int c = (int)(i % C);
    long r = i / C;
    int x2 = (int)(r % OW); r /= OW;
    int y2 = (int)(r % OH);
    int n  = (int)(r / OH);
    float v = X[(((long)n * C + c) * SH + (y2 >> 1)) * SW + (x2 >> 1)];
    v = fmaf(v, scale[c], shift[c]);
    Y[i] = v > 0.f ? v : 0.f;
}

// ---------------------------------------------------------------------------
// WMMA GEMM: C[M,N] = A[M,K]fp32 @ Wp(bf16 packed)  [+bias][+=C]
// full 32-k tiles: unconditional float4 loads; single tail tile: clamp+select
// ---------------------------------------------------------------------------
__global__ void __launch_bounds__(32)
gemm_bf16_wmma(const float* __restrict__ A, int K, int lda, int ktiles,
               const __bf16* __restrict__ Wp,
               const float* __restrict__ bias,
               float* __restrict__ C, int ldc, int accumulate) {
    const int tm = blockIdx.x * 16;
    const int tn = blockIdx.y * 16;
    const int lane = threadIdx.x & 31;
    const int l15 = lane & 15, half = lane >> 4;

    v8f acc;
#pragma unroll
    for (int j = 0; j < 8; ++j)
        acc[j] = accumulate ? C[(long)(tm + j + 8 * half) * ldc + tn + l15] : 0.f;

    const float* arow = A + (size_t)(tm + l15) * lda;
    const __bf16* wt = Wp + ((long)blockIdx.y * ktiles) * 512 + lane * 16;
    const int Kfull = K & ~31;

    int kt = 0;
    for (; kt * 32 < Kfull; ++kt) {
        __builtin_prefetch(arow + kt * 32 + 64, 0, 0);
        v16bf b = *(const v16bf*)(wt + (long)kt * 512);
        const float* p1 = arow + kt * 32 + half * 8;
        float4 q0 = *(const float4*)p1;
        float4 q1 = *(const float4*)(p1 + 4);
        float4 q2 = *(const float4*)(p1 + 16);
        float4 q3 = *(const float4*)(p1 + 20);
        v16bf a;
        a[0] = tobf(q0.x); a[1] = tobf(q0.y); a[2] = tobf(q0.z); a[3] = tobf(q0.w);
        a[4] = tobf(q1.x); a[5] = tobf(q1.y); a[6] = tobf(q1.z); a[7] = tobf(q1.w);
        a[8] = tobf(q2.x); a[9] = tobf(q2.y); a[10] = tobf(q2.z); a[11] = tobf(q2.w);
        a[12] = tobf(q3.x); a[13] = tobf(q3.y); a[14] = tobf(q3.z); a[15] = tobf(q3.w);
        acc = __builtin_amdgcn_wmma_f32_16x16x32_bf16(false, a, false, b,
                                                      (short)0, acc, false, false);
    }
    if (kt < ktiles) {                       // ragged tail: clamped loads, no branches
        v16bf b = *(const v16bf*)(wt + (long)kt * 512);
        v16bf a;
#pragma unroll
        for (int j = 0; j < 8; ++j) {
#pragma unroll
            for (int u = 0; u < 2; ++u) {
                int kb = kt * 32 + ((j < 4) ? 0 : 16) + half * 8 + (j & 3) * 2 + u;
                float v = arow[kb < K ? kb : 0];
                a[2 * j + u] = tobf(kb < K ? v : 0.f);
            }
        }
        acc = __builtin_amdgcn_wmma_f32_16x16x32_bf16(false, a, false, b,
                                                      (short)0, acc, false, false);
    }

#pragma unroll
    for (int j = 0; j < 8; ++j) {
        float v = acc[j];
        if (bias) v += bias[tn + l15];
        C[(long)(tm + j + 8 * half) * ldc + tn + l15] = v;
    }
}

// edge message GEMM on pre-padded EF (K multiple of 32): relu + atomic scatter to msum[dst]
__global__ void __launch_bounds__(32)
edge_gemm_scatter(const float* __restrict__ EF, int Kp, int ktiles,
                  const __bf16* __restrict__ Wp,
                  const float* __restrict__ bias, int N,
                  const int* __restrict__ dst, float* __restrict__ msum) {
    const int tm = blockIdx.x * 16;
    const int tn = blockIdx.y * 16;
    const int lane = threadIdx.x & 31;
    const int l15 = lane & 15, half = lane >> 4;

    v8f acc;
#pragma unroll
    for (int j = 0; j < 8; ++j) acc[j] = 0.f;

    const float* arow = EF + (size_t)(tm + l15) * Kp;
    const __bf16* wt = Wp + ((long)blockIdx.y * ktiles) * 512 + lane * 16;

    for (int kt = 0; kt < ktiles; ++kt) {
        v16bf b = *(const v16bf*)(wt + (long)kt * 512);
        const float* p1 = arow + kt * 32 + half * 8;
        float4 q0 = *(const float4*)p1;
        float4 q1 = *(const float4*)(p1 + 4);
        float4 q2 = *(const float4*)(p1 + 16);
        float4 q3 = *(const float4*)(p1 + 20);
        v16bf a;
        a[0] = tobf(q0.x); a[1] = tobf(q0.y); a[2] = tobf(q0.z); a[3] = tobf(q0.w);
        a[4] = tobf(q1.x); a[5] = tobf(q1.y); a[6] = tobf(q1.z); a[7] = tobf(q1.w);
        a[8] = tobf(q2.x); a[9] = tobf(q2.y); a[10] = tobf(q2.z); a[11] = tobf(q2.w);
        a[12] = tobf(q3.x); a[13] = tobf(q3.y); a[14] = tobf(q3.z); a[15] = tobf(q3.w);
        acc = __builtin_amdgcn_wmma_f32_16x16x32_bf16(false, a, false, b,
                                                      (short)0, acc, false, false);
    }

#pragma unroll
    for (int j = 0; j < 8; ++j) {
        int e = tm + j + 8 * half;
        float v = acc[j] + bias[tn + l15];
        v = v > 0.f ? v : 0.f;
        atomicAdd(&msum[(long)dst[e] * N + tn + l15], v);
    }
}

// ---------------------------------------------------------------------------
// implicit-GEMM conv, NHWC activations, K flattened over (tap, channel).
// Optional fused input BN+ReLU and 2x nearest upsample (upshift=1).
// Boundary/padding handled by clamped addresses + selects (branchless).
// ---------------------------------------------------------------------------
__global__ void __launch_bounds__(32)
conv_wmma(const float* __restrict__ In, int Cin, int cinShift, int SH, int SW, int upshift,
          const float* __restrict__ scale, const float* __restrict__ shift, int useBn,
          const __bf16* __restrict__ Wp, const float* __restrict__ bias,
          int KH, int KW, int ktiles,
          float* __restrict__ Out, int Cout, int OH, int OW,
          int accumulate, int outNCHW) {
    const int tm = blockIdx.x * 16;            // pixel tile (M = NODES*OH*OW)
    const int tn = blockIdx.y * 16;            // cout tile
    const int lane = threadIdx.x & 31;
    const int l15 = lane & 15, half = lane >> 4;
    const int HW = OH * OW;
    const int taps = KH * KW;
    const int ph = KH >> 1, pw = KW >> 1;

    const int pA = tm + l15;
    const int nN = pA / HW;
    const int remA = pA % HW;
    const int oy = remA / OW, ox = remA % OW;

    int oidx[8];
    v8f acc;
#pragma unroll
    for (int j = 0; j < 8; ++j) {
        int pC = tm + j + 8 * half;
        int n2 = pC / HW, r2 = pC % HW;
        int col = tn + l15;
        oidx[j] = outNCHW ? ((n2 * Cout + col) * OH + r2 / OW) * OW + (r2 % OW)
                          : (n2 * HW + r2) * Cout + col;
        acc[j] = accumulate ? Out[oidx[j]] : 0.f;
    }

    const __bf16* wt = Wp + ((long)blockIdx.y * ktiles) * 512 + lane * 16;
    const int cmask = Cin - 1;
    const int nbase = nN * SH * SW * Cin;

    for (int kt = 0; kt < ktiles; ++kt) {
        v16bf b = *(const v16bf*)(wt + (long)kt * 512);
        v16bf a;
#pragma unroll
        for (int ch = 0; ch < 2; ++ch) {
            int fs = kt * 32 + ch * 16 + half * 8;   // flat k of this 8-chunk
            int tap = fs >> cinShift;
            int cc = fs & cmask;
            int dy = tap / KW, dx = tap - (tap / KW) * KW;
            int iy = oy + dy - ph, ix = ox + dx - pw;
            bool ok = (tap < taps) & (iy >= 0) & (iy < OH) & (ix >= 0) & (ix < OW);
            int iyc = min(max(iy, 0), OH - 1);
            int ixc = min(max(ix, 0), OW - 1);
            const float* p = In + nbase + ((iyc >> upshift) * SW + (ixc >> upshift)) * Cin + cc;
            float4 q0 = *(const float4*)p;
            float4 q1 = *(const float4*)(p + 4);
            float vv[8] = {q0.x, q0.y, q0.z, q0.w, q1.x, q1.y, q1.z, q1.w};
            if (useBn) {
                float4 s0 = *(const float4*)(scale + cc);
                float4 s1 = *(const float4*)(scale + cc + 4);
                float4 t0 = *(const float4*)(shift + cc);
                float4 t1 = *(const float4*)(shift + cc + 4);
                float ss[8] = {s0.x, s0.y, s0.z, s0.w, s1.x, s1.y, s1.z, s1.w};
                float tt[8] = {t0.x, t0.y, t0.z, t0.w, t1.x, t1.y, t1.z, t1.w};
#pragma unroll
                for (int u = 0; u < 8; ++u) {
                    float v = fmaf(vv[u], ss[u], tt[u]);
                    vv[u] = v > 0.f ? v : 0.f;
                }
            }
#pragma unroll
            for (int u = 0; u < 8; ++u)
                a[ch * 8 + u] = tobf(ok ? vv[u] : 0.f);
        }
        acc = __builtin_amdgcn_wmma_f32_16x16x32_bf16(false, a, false, b,
                                                      (short)0, acc, false, false);
    }

#pragma unroll
    for (int j = 0; j < 8; ++j) {
        float v = acc[j];
        if (bias) v += bias[tn + l15];
        Out[oidx[j]] = v;
    }
}

// ---------------------------------------------------------------------------
// final: total_mask = segsum(masks, batch); layout = segsum(features*masks, batch)
// ---------------------------------------------------------------------------
__global__ void final_scatter(const float* __restrict__ masks, const int* __restrict__ batch,
                              const float* __restrict__ feat,
                              float* __restrict__ layout, float* __restrict__ total) {
    long i = (long)blockIdx.x * blockDim.x + threadIdx.x;
    if (i >= (long)NODES * 16 * 4096) return;
    int n = (int)(i >> 16);
    int r = (int)(i & 65535);
    int c = r >> 12;
    int b = batch[n];
    float m = masks[i];
    atomicAdd(&total[b * 65536 + r], m);
    atomicAdd(&layout[b * 65536 + r], feat[n * 16 + c] * m);
}

// ---------------------------------------------------------------------------
// host orchestration
// ---------------------------------------------------------------------------
static inline unsigned nb(long n, int t = 256) { return (unsigned)((n + t - 1) / t); }

extern "C" void kernel_launch(void* const* d_in, const int* in_sizes, int n_in,
                              void* d_out, int out_size, void* d_ws, size_t ws_size,
                              hipStream_t stream) {
    (void)in_sizes; (void)n_in; (void)out_size; (void)ws_size;

    const int*   xI    = (const int*)  d_in[0];
    const float* pos   = (const float*)d_in[1];
    const int*   src   = (const int*)  d_in[2];
    const int*   dst   = src + EDGES;
    const int*   batch = (const int*)  d_in[3];
    const float* emb   = (const float*)d_in[4];
    const float* f0p[5]  = {(const float*)d_in[5],  (const float*)d_in[6],  (const float*)d_in[7],  (const float*)d_in[8],  (const float*)d_in[9]};
    const float* f0bn[2] = {(const float*)d_in[10], (const float*)d_in[11]};
    const float* f1p[5]  = {(const float*)d_in[12], (const float*)d_in[13], (const float*)d_in[14], (const float*)d_in[15], (const float*)d_in[16]};
    const float* f1bn[2] = {(const float*)d_in[17], (const float*)d_in[18]};
    const float* f2p[5]  = {(const float*)d_in[19], (const float*)d_in[20], (const float*)d_in[21], (const float*)d_in[22], (const float*)d_in[23]};
    const float* m0p[5]  = {(const float*)d_in[24], (const float*)d_in[25], (const float*)d_in[26], (const float*)d_in[27], (const float*)d_in[28]};
    const float* m0bn[2] = {(const float*)d_in[29], (const float*)d_in[30]};
    const float* m1p[5]  = {(const float*)d_in[31], (const float*)d_in[32], (const float*)d_in[33], (const float*)d_in[34], (const float*)d_in[35]};
    const float* m1bn[2] = {(const float*)d_in[36], (const float*)d_in[37]};
    const float* m2p[5]  = {(const float*)d_in[38], (const float*)d_in[39], (const float*)d_in[40], (const float*)d_in[41], (const float*)d_in[42]};
    const float* u0bn[2] = {(const float*)d_in[43], (const float*)d_in[44]};
    const float* u0_w  = (const float*)d_in[45];
    const float* u0_b  = (const float*)d_in[46];
    const float* u0_wn = (const float*)d_in[47];
    const float* u0_b2 = (const float*)d_in[48];
    const float* u1bn[2] = {(const float*)d_in[49], (const float*)d_in[50]};
    const float* u1_w  = (const float*)d_in[51];
    const float* u1_b  = (const float*)d_in[52];
    const float* u2bn[2] = {(const float*)d_in[53], (const float*)d_in[54]};
    const float* u2_w  = (const float*)d_in[55];
    const float* u2_b  = (const float*)d_in[56];
    const float* u3bn[2] = {(const float*)d_in[57], (const float*)d_in[58]};
    const float* u3_w  = (const float*)d_in[59];
    const float* u3_b  = (const float*)d_in[60];
    const float* obn[2]  = {(const float*)d_in[61], (const float*)d_in[62]};
    const float* out_w = (const float*)d_in[63];
    const float* out_b = (const float*)d_in[64];

    float* layoutO = (float*)d_out;
    float* totalm  = layoutO + (long)NGRAPHS * 16 * 64 * 64;
    float* masks   = totalm + (long)NGRAPHS * 16 * 64 * 64;     // NCHW [1024,16,64,64]

    float* ws = (float*)d_ws;
    float* EMB   = ws;                          // 65536
    float* nA    = EMB   + 65536;               // 1024*2048
    float* nB    = nA    + 2097152;             // 1024*2048
    float* msum  = nB    + 2097152;             // 1024*2048
    float* EF    = msum  + 2097152;             // 8192*160 (padded)
    float* wpkF  = EF    + 1310720;             // 2097152 floats -> 4.19M bf16
    float* deg   = wpkF  + 2097152;             // 1024
    float* feat  = deg   + 1024;                // 1024*16
    float* mu    = feat  + 16384;               // 2048
    float* var   = mu    + 2048;
    float* scale = var   + 2048;
    float* shift = scale + 2048;
    float* hh    = shift + 2048;                // NHWC [1024,8,8,128]
    float* aggr  = hh    + 8388608;             // NHWC [1024,8,8,128]
    float* y0    = aggr  + 8388608;             // NHWC [1024,8,8,64]
    float* y1    = y0    + 4194304;             // NHWC [1024,16,16,32]
    float* y2    = y1    + 8388608;             // NHWC [1024,32,32,16]
    float* y3    = y2    + 16777216;            // NHWC [1024,64,64,16]
    __bf16* wpk  = (__bf16*)wpkF;

    zero_f32<<<nb(NODES), 256, 0, stream>>>(deg, NODES);
    count_deg<<<nb(EDGES), 256, 0, stream>>>(dst, deg);
    gather_emb<<<nb(NODES * 64), 256, 0, stream>>>(xI, emb, EMB);

    auto gemm = [&](const float* A, int K, int lda, const float* W, int N,
                    const float* bias, float* C, int accum) {
        int kt = (K + 31) / 32;
        pack_w_gemm<<<nb((long)(N / 16) * kt * 512), 256, 0, stream>>>(W, K, N, kt, wpk);
        dim3 g(NODES / 16, N / 16);
        gemm_bf16_wmma<<<g, 32, 0, stream>>>(A, K, lda, kt, wpk, bias, C, N, accum);
    };

    auto gpose = [&](const float* X, int Cin, int hid, int cout,
                     const float* const* P, float* out) {
        int K = Cin + 3, Kp = (K + 31) & ~31, ktE = Kp / 32;
        build_ef<<<nb((long)EDGES * Kp), 256, 0, stream>>>(X, Cin, Kp, pos, src, dst, EF);
        pack_w_gemm<<<nb((long)(hid / 16) * ktE * 512), 256, 0, stream>>>(P[0], K, hid, ktE, wpk);
        zero_f32<<<nb((long)NODES * hid), 256, 0, stream>>>(msum, (long)NODES * hid);
        dim3 ge(EDGES / 16, hid / 16);
        edge_gemm_scatter<<<ge, 32, 0, stream>>>(EF, Kp, ktE, wpk, P[1], hid, dst, msum);
        div_rows<<<nb((long)NODES * hid), 256, 0, stream>>>(msum, deg, (long)NODES * hid, hid);
        gemm(X, Cin, Cin, P[2], cout, P[3], out, 0);
        gemm(msum, hid, hid, P[2] + (long)Cin * cout, cout, nullptr, out, 1);
        gemm(X, Cin, Cin, P[4], cout, nullptr, out, 1);
    };
    auto bn1d = [&](float* X, int C, const float* const* gb) {
        bn1d_stats<<<C, 256, 0, stream>>>(X, C, mu, var);
        bn1d_apply<<<nb(NODES * C), 256, 0, stream>>>(X, C, mu, var, gb[0], gb[1]);
    };
    auto bn2d_ss = [&](const float* X, int C, int S, int nhwc, const float* const* gb) {
        bn2d_stats<<<C, 256, 0, stream>>>(X, C, S, nhwc, mu, var);
        make_scale_shift<<<1, 256, 0, stream>>>(mu, var, gb[0], gb[1], scale, shift, C);
    };
    auto conv = [&](const float* In, int Cin, int cinShift, int SH, int SW, int up2,
                    int useBn, const float* Wt, const float* bias, int KH, int KW,
                    float* Out, int Cout, int OH, int OW, int accum, int nchwOut) {
        int kt = (KH * KW * Cin + 31) / 32;
        pack_w_conv<<<nb((long)(Cout / 16) * kt * 512), 256, 0, stream>>>(
            Wt, Cin, cinShift, KH, KW, Cout, kt, wpk);
        dim3 g((unsigned)((long)NODES * OH * OW / 16), Cout / 16);
        conv_wmma<<<g, 32, 0, stream>>>(In, Cin, cinShift, SH, SW, up2,
                                        scale, shift, useBn, wpk, bias, KH, KW, kt,
                                        Out, Cout, OH, OW, accum, nchwOut);
    };

    // ---- feature extractor -> features [N,16] ----
    gpose(EMB, 64, 64, 64, f0p, nA);  bn1d(nA, 64, f0bn);
    gpose(nA, 64, 32, 32, f1p, nB);   bn1d(nB, 32, f1bn);
    gpose(nB, 32, 16, 16, f2p, feat);

    // ---- mask branch graph features -> g [N,128,4,4] (nA, NCHW) ----
    gpose(pos, 3, 64, 64, m0p, nA);    bn1d(nA, 64, m0bn);
    gpose(nA, 64, 128, 128, m1p, nB);  bn1d(nB, 128, m1bn);
    gpose(nB, 128, 2048, 2048, m2p, nA);

    // ---- PG2DBlock ----
    bn2d_ss(nA, 128, 16, 0, u0bn);
    up2_bn_relu<<<nb((long)NODES * 128 * 64), 256, 0, stream>>>(nA, hh, 128, 4, 4, scale, shift);
    conv(hh, 128, 7, 8, 8, 0, 0, u0_w, u0_b, 3, 3, y0, 64, 8, 8, 0, 0);
    zero_f32<<<nb((long)NODES * 8192), 256, 0, stream>>>(aggr, (long)NODES * 8192);
    edge_scatter_vec<<<nb((long)EDGES * 8192), 256, 0, stream>>>(src, dst, hh, aggr, 8192);
    div_rows<<<nb((long)NODES * 8192), 256, 0, stream>>>(aggr, deg, (long)NODES * 8192, 8192);
    conv(aggr, 128, 7, 8, 8, 0, 0, u0_wn, u0_b2, 1, 1, y0, 64, 8, 8, 1, 0);

    // ---- GBlocks: bn-relu-up2 fused into conv A-load ----
    bn2d_ss(y0, 64, 64, 1, u1bn);
    conv(y0, 64, 6, 8, 8, 1, 1, u1_w, u1_b, 3, 3, y1, 32, 16, 16, 0, 0);
    bn2d_ss(y1, 32, 256, 1, u2bn);
    conv(y1, 32, 5, 16, 16, 1, 1, u2_w, u2_b, 3, 3, y2, 16, 32, 32, 0, 0);
    bn2d_ss(y2, 16, 1024, 1, u3bn);
    conv(y2, 16, 4, 32, 32, 1, 1, u3_w, u3_b, 3, 3, y3, 16, 64, 64, 0, 0);

    // ---- output conv -> masks (NCHW in d_out) ----
    bn2d_ss(y3, 16, 4096, 1, obn);
    conv(y3, 16, 4, 64, 64, 0, 1, out_w, out_b, 3, 3, masks, 16, 64, 64, 0, 1);

    // ---- segment reductions ----
    zero_f32<<<nb((long)2 * NGRAPHS * 16 * 64 * 64), 256, 0, stream>>>(
        layoutO, (long)2 * NGRAPHS * 16 * 64 * 64);
    final_scatter<<<nb((long)NODES * 16 * 4096), 256, 0, stream>>>(masks, batch, feat,
                                                                   layoutO, totalm);
}
